// LSTM1_5987184410628
// MI455X (gfx1250) — compile-verified, matrix operand order checked by reading, not run on hardware
//
#include <hip/hip_runtime.h>

// ---------------------------------------------------------------------------
// Persistent 2-layer LSTM for MI455X (gfx1250):
//   - bf16 WMMA (v_wmma_f32_16x16x32_bf16), cell state in registers
//   - per-block weight tiles resident in 320KB LDS (ds_load_b128 fragments)
//   - 2 device-wide barriers per timestep, linear head fused via shfl+atomics
// ---------------------------------------------------------------------------

typedef __attribute__((ext_vector_type(16))) __bf16 v16bf;
typedef __attribute__((ext_vector_type(8)))  float  v8f;

#define HDIM 512      // hidden size
#define BATCH 256     // batch
#define NBLK 64       // persistent blocks: 64 x 256 thr = 512 waves = 512 tiles

#define RS    1040            // LDS row stride bytes (1024 payload + 16 pad)
#define TROWS (16 * RS)       // bytes per 16-row weight tile in LDS
#define SMEM_BYTES (12 * TROWS)  // 12 tiles (3 matrices x 4 gates) = 199,680 B

// ---------------- helpers ----------------
__device__ __forceinline__ unsigned short f2bf(float x) {
  unsigned u = __float_as_uint(x);
  u += 0x7FFFu + ((u >> 16) & 1u);          // round-to-nearest-even
  return (unsigned short)(u >> 16);
}

__device__ __forceinline__ float fsig(float x) { return 1.f / (1.f + __expf(-x)); }
__device__ __forceinline__ float ftanh(float x) {
  x = fminf(15.f, fmaxf(-15.f, x));
  float e = __expf(2.f * x);
  return (e - 1.f) / (e + 1.f);
}

__device__ __forceinline__ v8f vzero8() {
  v8f z;
#pragma unroll
  for (int i = 0; i < 8; ++i) z[i] = 0.f;
  return z;
}

// 16-bit A/B fragment for V_WMMA_*_16X16X32: lane L holds row (L&15);
// lanes 0-15: K = koff+0..7 (VGPR0-3) and koff+16..23 (VGPR4-7);
// lanes 16-31: same pattern at koff+8.  Two b128 loads per fragment.
union FragU { v16bf v; uint4 q[2]; };

__device__ __forceinline__ v16bf load_frag_g(const unsigned short* rowp, int koff) {
  FragU f;
  const uint4* p = (const uint4*)(rowp + koff);
  f.q[0] = p[0];   // elements koff .. koff+7
  f.q[1] = p[2];   // elements koff+16 .. koff+23
  return f.v;
}

__device__ __forceinline__ v16bf load_frag_l(const char* lanerow, int koff) {
  FragU f;
  const uint4* p = (const uint4*)(lanerow + (koff << 1));
  f.q[0] = p[0];
  f.q[1] = p[2];
  return f.v;
}

__device__ __forceinline__ v8f wmma_bf16(v16bf a, v16bf b, v8f c) {
  return __builtin_amdgcn_wmma_f32_16x16x32_bf16(false, a, false, b, (short)0, c,
                                                 false, false);
}

// sense-reversing grid barrier, agent scope
__device__ __forceinline__ void gsync(unsigned* cnt, unsigned* gen, unsigned nb) {
  __builtin_amdgcn_fence(__ATOMIC_RELEASE, "agent");
  __syncthreads();
  if (threadIdx.x == 0) {
    unsigned g = __hip_atomic_load(gen, __ATOMIC_RELAXED, __HIP_MEMORY_SCOPE_AGENT);
    unsigned old = __hip_atomic_fetch_add(cnt, 1u, __ATOMIC_ACQ_REL, __HIP_MEMORY_SCOPE_AGENT);
    if (old == nb - 1u) {
      __hip_atomic_store(cnt, 0u, __ATOMIC_RELAXED, __HIP_MEMORY_SCOPE_AGENT);
      __hip_atomic_fetch_add(gen, 1u, __ATOMIC_ACQ_REL, __HIP_MEMORY_SCOPE_AGENT);
    } else {
      while (__hip_atomic_load(gen, __ATOMIC_RELAXED, __HIP_MEMORY_SCOPE_AGENT) == g) {
        __builtin_amdgcn_s_sleep(1);
      }
    }
  }
  __syncthreads();
  __builtin_amdgcn_fence(__ATOMIC_ACQUIRE, "agent");
}

// ---------------- prep: fp32 -> bf16 weights, zero state ----------------
__global__ void lstm_prep(const float* __restrict__ Whh1,
                          const float* __restrict__ Wih2,
                          const float* __restrict__ Whh2,
                          unsigned short* wbase,       // 3 * 4H*H bf16
                          unsigned short* hbase,       // 4 * B*H bf16
                          float* accbase,              // 2 * BATCH f32
                          unsigned* bar,
                          int WN, int HB4) {
  int i = blockIdx.x * blockDim.x + threadIdx.x;
  int stride = gridDim.x * blockDim.x;
  for (int k = i; k < WN; k += stride) {
    wbase[k]          = f2bf(Whh1[k]);
    wbase[WN + k]     = f2bf(Wih2[k]);
    wbase[2 * WN + k] = f2bf(Whh2[k]);
  }
  for (int k = i; k < HB4; k += stride) hbase[k] = 0;
  if (i < 2 * BATCH) accbase[i] = 0.f;
  if (i < 2) bar[i] = 0u;
}

// ---------------- persistent LSTM ----------------
__global__ __launch_bounds__(256)
void lstm_persistent(const float* __restrict__ input_t,
                     const float* __restrict__ W_ih1,
                     const float* __restrict__ b_ih1,
                     const float* __restrict__ b_hh1,
                     const float* __restrict__ b_ih2,
                     const float* __restrict__ b_hh2,
                     const float* __restrict__ W_lin,
                     const float* __restrict__ b_lin,
                     const unsigned short* __restrict__ whh1,
                     const unsigned short* __restrict__ wih2,
                     const unsigned short* __restrict__ whh2,
                     unsigned short* h1a, unsigned short* h1b,
                     unsigned short* h2a, unsigned short* h2b,
                     float* accbase, unsigned* bar, float* dout,
                     int T, int TT) {
  extern __shared__ char smem[];

  const int lane = threadIdx.x & 31;
  const int wv   = threadIdx.x >> 5;            // wave in block (0..7)
  const int jt   = blockIdx.x & 31;             // hidden-column tile (block-shared)
  const int bt   = (blockIdx.x >> 5) * 8 + wv;  // batch tile 0..15
  const int l15  = lane & 15;
  const int kb   = (lane & 16) ? 8 : 0;         // K sub-offset for fragments
  const int mhi  = (lane & 16) ? 8 : 0;         // C/D row offset
  const int nidx = jt * 16 + l15;               // hidden column owned by lane
  const int rA   = bt * 16 + l15;               // batch row for A fragments

  // ---- stage this block's 12 weight tiles (192 rows x 1KB) into LDS ----
  for (int idx = threadIdx.x; idx < 192 * 64; idx += 256) {
    int row = idx >> 6;                 // 0..191
    int q   = idx & 63;                 // 16-byte chunk within row
    int m   = row >> 6;                 // 0: Whh1, 1: Wih2, 2: Whh2
    int sub = row & 63;                 // gate*16 + r
    const unsigned short* src = (m == 0) ? whh1 : ((m == 1) ? wih2 : whh2);
    int grow = ((sub >> 4) * HDIM) + jt * 16 + (sub & 15);   // g*H + jt*16 + r
    const uint4* s4 = (const uint4*)(src + (size_t)grow * HDIM) + q;
    *(uint4*)(smem + row * RS + q * 16) = *s4;
  }
  __syncthreads();

  // per-lane LDS row base for B fragments (row = l15 within each tile)
  const char* lrow = smem + l15 * RS;
  const char* t1g0 = lrow + 0 * TROWS;   // layer1 Whh1 gates
  const char* t1g1 = lrow + 1 * TROWS;
  const char* t1g2 = lrow + 2 * TROWS;
  const char* t1g3 = lrow + 3 * TROWS;
  const char* tag0 = lrow + 4 * TROWS;   // layer2 Wih2 gates
  const char* tag1 = lrow + 5 * TROWS;
  const char* tag2 = lrow + 6 * TROWS;
  const char* tag3 = lrow + 7 * TROWS;
  const char* tbg0 = lrow + 8 * TROWS;   // layer2 Whh2 gates
  const char* tbg1 = lrow + 9 * TROWS;
  const char* tbg2 = lrow + 10 * TROWS;
  const char* tbg3 = lrow + 11 * TROWS;

  // folded scalars (constant across steps)
  const float wx0 = W_ih1[0 * HDIM + nidx], wx1 = W_ih1[1 * HDIM + nidx];
  const float wx2 = W_ih1[2 * HDIM + nidx], wx3 = W_ih1[3 * HDIM + nidx];
  const float bs10 = b_ih1[0 * HDIM + nidx] + b_hh1[0 * HDIM + nidx];
  const float bs11 = b_ih1[1 * HDIM + nidx] + b_hh1[1 * HDIM + nidx];
  const float bs12 = b_ih1[2 * HDIM + nidx] + b_hh1[2 * HDIM + nidx];
  const float bs13 = b_ih1[3 * HDIM + nidx] + b_hh1[3 * HDIM + nidx];
  const float bs20 = b_ih2[0 * HDIM + nidx] + b_hh2[0 * HDIM + nidx];
  const float bs21 = b_ih2[1 * HDIM + nidx] + b_hh2[1 * HDIM + nidx];
  const float bs22 = b_ih2[2 * HDIM + nidx] + b_hh2[2 * HDIM + nidx];
  const float bs23 = b_ih2[3 * HDIM + nidx] + b_hh2[3 * HDIM + nidx];
  const float wl = W_lin[nidx];
  const float bl = b_lin[0];

  v8f c1 = vzero8();   // persistent cell state, in registers for entire run
  v8f c2 = vzero8();

  unsigned* cnt = bar;
  unsigned* gen = bar + 1;
  const unsigned nb = gridDim.x;

  for (int t = 0; t < TT; ++t) {
    const unsigned short* h1_rd = (t & 1) ? h1b : h1a;
    unsigned short*       h1_wr = (t & 1) ? h1a : h1b;
    const unsigned short* h2_rd = (t & 1) ? h2b : h2a;
    unsigned short*       h2_wr = (t & 1) ? h2a : h2b;
    const float* acc_prev = accbase + ((t - 1) & 1) * BATCH;  // out(t-1) partials
    float*       acc_wr   = accbase + (t & 1) * BATCH;

    // ---------- phase 1: layer 1 (+ emit d_out for step t-1) ----------
    if (blockIdx.x == 0 && t > 0) {
      int b0 = threadIdx.x;
      dout[(size_t)b0 * TT + (t - 1)] = acc_prev[b0] + bl;
    }
    {
      const unsigned short* hr = h1_rd + (size_t)rA * HDIM;
      v8f a0 = vzero8(), a1 = vzero8(), a2 = vzero8(), a3 = vzero8();
#pragma unroll 2
      for (int k = 0; k < HDIM; k += 32) {
        int ko = k + kb;
        v16bf av = load_frag_g(hr, ko);
        a0 = wmma_bf16(av, load_frag_l(t1g0, ko), a0);
        a1 = wmma_bf16(av, load_frag_l(t1g1, ko), a1);
        a2 = wmma_bf16(av, load_frag_l(t1g2, ko), a2);
        a3 = wmma_bf16(av, load_frag_l(t1g3, ko), a3);
      }
#pragma unroll
      for (int v = 0; v < 8; ++v) {
        int b = bt * 16 + mhi + v;
        float xv = (t < T) ? input_t[b] : (acc_prev[b] + bl);
        float gi = fsig (a0[v] + xv * wx0 + bs10);
        float gf = fsig (a1[v] + xv * wx1 + bs11);
        float gg = ftanh(a2[v] + xv * wx2 + bs12);
        float go = fsig (a3[v] + xv * wx3 + bs13);
        float c = gf * c1[v] + gi * gg;
        c1[v] = c;
        h1_wr[(size_t)b * HDIM + nidx] = f2bf(go * ftanh(c));
      }
    }
    gsync(cnt, gen, nb);

    // ---------- phase 2: layer 2 + fused linear-head partials ----------
    if (blockIdx.x == 0 && t > 0) {
      // retire the accumulator consumed in phase 1 (reused at step t+1)
      accbase[((t - 1) & 1) * BATCH + threadIdx.x] = 0.f;
    }
    {
      const unsigned short* hr1 = h1_wr + (size_t)rA * HDIM;  // new h1
      const unsigned short* hr2 = h2_rd + (size_t)rA * HDIM;  // previous h2
      v8f a0 = vzero8(), a1 = vzero8(), a2 = vzero8(), a3 = vzero8();
#pragma unroll 2
      for (int k = 0; k < HDIM; k += 32) {
        int ko = k + kb;
        v16bf av = load_frag_g(hr1, ko);
        a0 = wmma_bf16(av, load_frag_l(tag0, ko), a0);
        a1 = wmma_bf16(av, load_frag_l(tag1, ko), a1);
        a2 = wmma_bf16(av, load_frag_l(tag2, ko), a2);
        a3 = wmma_bf16(av, load_frag_l(tag3, ko), a3);
        v16bf hv = load_frag_g(hr2, ko);
        a0 = wmma_bf16(hv, load_frag_l(tbg0, ko), a0);
        a1 = wmma_bf16(hv, load_frag_l(tbg1, ko), a1);
        a2 = wmma_bf16(hv, load_frag_l(tbg2, ko), a2);
        a3 = wmma_bf16(hv, load_frag_l(tbg3, ko), a3);
      }
#pragma unroll
      for (int v = 0; v < 8; ++v) {
        int b = bt * 16 + mhi + v;
        float gi = fsig (a0[v] + bs20);
        float gf = fsig (a1[v] + bs21);
        float gg = ftanh(a2[v] + bs22);
        float go = fsig (a3[v] + bs23);
        float c = gf * c2[v] + gi * gg;
        c2[v] = c;
        float hv2 = go * ftanh(c);
        h2_wr[(size_t)b * HDIM + nidx] = f2bf(hv2);
        // linear head partial: reduce 16 columns of this tile row via shuffles
        float s = hv2 * wl;
        s += __shfl_xor(s, 1);
        s += __shfl_xor(s, 2);
        s += __shfl_xor(s, 4);
        s += __shfl_xor(s, 8);
        if (l15 == 0) atomicAdd(&acc_wr[b], s);
      }
    }
    gsync(cnt, gen, nb);
  }

  // final output column (step TT-1)
  if (blockIdx.x == 0) {
    int b0 = threadIdx.x;
    dout[(size_t)b0 * TT + (TT - 1)] = accbase[((TT - 1) & 1) * BATCH + b0] + bl;
  }
}

// ---------------------------------------------------------------------------
extern "C" void kernel_launch(void* const* d_in, const int* in_sizes, int n_in,
                              void* d_out, int out_size, void* d_ws, size_t ws_size,
                              hipStream_t stream) {
  const float* input_t = (const float*)d_in[0];
  // d_in[1] = y (only defines T)
  const float* W_ih1 = (const float*)d_in[2];
  const float* b_ih1 = (const float*)d_in[3];
  const float* W_hh1 = (const float*)d_in[4];
  const float* b_hh1 = (const float*)d_in[5];
  const float* W_ih2 = (const float*)d_in[6];
  const float* b_ih2 = (const float*)d_in[7];
  const float* W_hh2 = (const float*)d_in[8];
  const float* b_hh2 = (const float*)d_in[9];
  const float* W_lin = (const float*)d_in[10];
  const float* b_lin = (const float*)d_in[11];
  // d_in[12] = future_preds (device scalar; derived from out_size instead)

  const int B = BATCH, H = HDIM;
  const int T  = in_sizes[1] / B;   // 512
  const int TT = out_size / B;      // T + future_preds

  char* ws = (char*)d_ws;
  unsigned* bar         = (unsigned*)ws;                 // 2 u32
  float* accbase        = (float*)(ws + 256);            // 2 * B floats
  unsigned short* hbase = (unsigned short*)(ws + 4096);  // 4 * B*H bf16
  unsigned short* h1a = hbase;
  unsigned short* h1b = hbase + (size_t)B * H;
  unsigned short* h2a = hbase + (size_t)2 * B * H;
  unsigned short* h2b = hbase + (size_t)3 * B * H;
  unsigned short* wbase = (unsigned short*)(ws + 4096 + (size_t)4 * B * H * 2);
  const int WN = 4 * H * H;                              // elems per weight matrix
  unsigned short* whh1 = wbase;
  unsigned short* wih2 = wbase + (size_t)WN;
  unsigned short* whh2 = wbase + (size_t)2 * WN;

  // allow ~195KB dynamic LDS (WGP has 320KB on gfx1250)
  static bool attr_done = false;
  if (!attr_done) {
    (void)hipFuncSetAttribute((const void*)lstm_persistent,
                              hipFuncAttributeMaxDynamicSharedMemorySize,
                              SMEM_BYTES);
    attr_done = true;
  }

  lstm_prep<<<1024, 256, 0, stream>>>(W_hh1, W_ih2, W_hh2, wbase, hbase,
                                      accbase, bar, WN, 4 * B * H);

  lstm_persistent<<<NBLK, 256, SMEM_BYTES, stream>>>(
      input_t, W_ih1, b_ih1, b_hh1, b_ih2, b_hh2, W_lin, b_lin,
      whh1, wih2, whh2, h1a, h1b, h2a, h2b,
      accbase, bar, (float*)d_out, T, TT);
}
